// BackwardTransform_87875030876802
// MI455X (gfx1250) — matrix operand order repaired and checked by reading, not compiled, not run on hardware
//
#include <hip/hip_runtime.h>
#include <hip/hip_bf16.h>

typedef __attribute__((ext_vector_type(16))) __bf16 v16bf;
typedef __attribute__((ext_vector_type(8)))  __bf16 bf16x8;
typedef __attribute__((ext_vector_type(8)))  float  v8f;
typedef int async_v4i __attribute__((vector_size(16)));   // element type the async builtin wants

#define CIN_   256
#define COUT_  80
#define NIMG_  6
#define FH_    32
#define FW_    88
#define PIX_   (FH_ * FW_)     // 2816
#define K3_    (CIN_ * 9)      // 2304
#define NX_    128
#define NY_    128
#define NZ_    10
#define NPTS_  (NX_ * NY_ * NZ_)  // 163840
#define IMGW_  704.0f
#define IMGH_  256.0f

// Async global->LDS (CDNA5): use if this toolchain exposes the builtins.
#if defined(__has_builtin)
#if __has_builtin(__builtin_amdgcn_global_load_async_to_lds_b128) && \
    __has_builtin(__builtin_amdgcn_s_wait_asynccnt)
#define USE_ASYNC_LDS 1
#endif
#endif

// ---------- helpers ----------
static __device__ __forceinline__ unsigned short f2bf(float f) {
  unsigned int u = __float_as_uint(f);
  u += 0x7FFFu + ((u >> 16) & 1u);          // round-to-nearest-even
  return (unsigned short)(u >> 16);
}

// ---------- elementwise f32 -> bf16 ----------
__global__ void cvt_bf16_kernel(const float* __restrict__ src,
                                unsigned short* __restrict__ dst, int n) {
  int i = blockIdx.x * blockDim.x + threadIdx.x;
  if (i < n) dst[i] = f2bf(src[i]);
}

// ---------- pack 3x3 weights OIHW -> [O][k], k = (kh*3+kw)*256 + c ----------
__global__ void pack_w3x3_kernel(const float* __restrict__ w,
                                 unsigned short* __restrict__ dst) {
  int i = blockIdx.x * blockDim.x + threadIdx.x;
  if (i >= CIN_ * K3_) return;
  int o = i / K3_, k = i % K3_;
  int tap = k / CIN_, c = k % CIN_;
  int kh = tap / 3, kw = tap % 3;
  dst[i] = f2bf(w[((o * CIN_ + c) * 3 + kh) * 3 + kw]);
}

// ---------- 3x3 conv + BN + ReLU as implicit-GEMM WMMA ----------
// block = 256 thr (8 waves) computes 64(M) x 64(N); each wave: 1 A-frag, 2 WMMAs/K-step.
#define ASTRIDE 40   // padded LDS row strides (bank spread), keep 16B/32B alignment
#define BSTRIDE 80

__global__ __launch_bounds__(256)
void conv3x3_bn_relu_wmma(const unsigned short* __restrict__ X,   // [NIMG][CIN][PIX] bf16
                          const unsigned short* __restrict__ Wa,  // [CIN][K3] bf16 packed
                          const float* __restrict__ bias,
                          const float* __restrict__ gam,
                          const float* __restrict__ bet,
                          const float* __restrict__ mu,
                          const float* __restrict__ var,
                          unsigned short* __restrict__ Y) {       // [NIMG][CIN][PIX] bf16
  __shared__ __align__(16) unsigned short ldsA[64 * ASTRIDE];
  __shared__ __align__(32) unsigned short ldsB[32 * BSTRIDE];

  const int img    = blockIdx.z;
  const int Mblock = blockIdx.y * 64;
  const int pbase  = blockIdx.x * 64;
  const int t      = threadIdx.x;
  const int lane   = t & 31;
  const int wid    = t >> 5;
  const int Moff   = (wid >> 1) * 16;          // wave's M tile (0..3)
  const int Nbase  = (wid & 1) * 32;           // wave's pair of N tiles
  const int m      = lane & 15;
  const int hh     = lane >> 4;

  const unsigned short* Xi = X + (size_t)img * (CIN_ * PIX_);

  // ---- B staging assignment: row kk = t>>3 (k within 32-group), 8 cols from nb ----
  const int kk = t >> 3;
  const int nb = (t & 7) * 8;
  // pixel decomposition is invariant across the whole K loop: hoist it.
  int ph[8], pw[8];
#pragma unroll
  for (int j = 0; j < 8; ++j) {
    int p = pbase + nb + j;
    ph[j] = p / FW_;
    pw[j] = p - ph[j] * FW_;
  }

  // ---- A staging assignment: one b128 per thread ----
  const int arow = t >> 2, achunk = (t & 3) * 8;
  const unsigned short* Asrc = Wa + (size_t)(Mblock + arow) * K3_ + achunk;

  v8f acc0 = {}, acc1 = {};

  for (int tap = 0; tap < 9; ++tap) {
    const int dh = tap / 3 - 1;
    const int dw = tap % 3 - 1;
    // per-tap, per-element source offset + validity (branch-free, reused for 8 c-groups)
    int   off[8];
    bool  val[8];
#pragma unroll
    for (int j = 0; j < 8; ++j) {
      int h2 = ph[j] + dh, w2 = pw[j] + dw;
      val[j] = ((unsigned)h2 < (unsigned)FH_) && ((unsigned)w2 < (unsigned)FW_);
      off[j] = val[j] ? (h2 * FW_ + w2) : 0;
    }

    for (int cg = 0; cg < 8; ++cg) {
      const int kbase = tap * 256 + cg * 32;

      // ---- stage A tile (64x32 weights) ----
#ifdef USE_ASYNC_LDS
      __builtin_amdgcn_global_load_async_to_lds_b128(
          (__attribute__((address_space(1))) async_v4i*)(Asrc + kbase),
          (__attribute__((address_space(3))) async_v4i*)&ldsA[arow * ASTRIDE + achunk],
          0, 0);
#else
      *(uint4*)&ldsA[arow * ASTRIDE + achunk] = *(const uint4*)(Asrc + kbase);
#endif
      if (cg + 1 < 8)
        __builtin_prefetch(Asrc + kbase + 32, 0, 1);

      // ---- stage B tile (32k x 64 pixels), unconditional gather + select ----
      {
        const unsigned short* Xc = Xi + (size_t)(cg * 32 + kk) * PIX_;
        unsigned short e[8];
#pragma unroll
        for (int j = 0; j < 8; ++j) {
          unsigned short x = Xc[off[j]];
          e[j] = val[j] ? x : (unsigned short)0;
        }
        uint4 pk;
        pk.x = (unsigned)e[0] | ((unsigned)e[1] << 16);
        pk.y = (unsigned)e[2] | ((unsigned)e[3] << 16);
        pk.z = (unsigned)e[4] | ((unsigned)e[5] << 16);
        pk.w = (unsigned)e[6] | ((unsigned)e[7] << 16);
        *(uint4*)&ldsB[kk * BSTRIDE + nb] = pk;
      }

#ifdef USE_ASYNC_LDS
      __builtin_amdgcn_s_wait_asynccnt(0);
#endif
      __syncthreads();

      // A fragment: lane m = L&15, half = L>>4; VGPR0-3: k=8h..8h+7, VGPR4-7: k=16+8h..
      v16bf afrag;
      ((bf16x8*)&afrag)[0] = *(const bf16x8*)&ldsA[(Moff + m) * ASTRIDE + 8 * hh];
      ((bf16x8*)&afrag)[1] = *(const bf16x8*)&ldsA[(Moff + m) * ASTRIDE + 16 + 8 * hh];
      // B fragments: lane L holds column n=L&15, k in same order as A (8h.., 16+8h..)
      v16bf b0, b1;
#pragma unroll
      for (int s = 0; s < 16; ++s) {
        int kr = (s < 8) ? (8 * hh + s) : (16 + 8 * hh + (s - 8));
        b0[s] = ((const __bf16*)ldsB)[kr * BSTRIDE + Nbase + m];
        b1[s] = ((const __bf16*)ldsB)[kr * BSTRIDE + Nbase + 16 + m];
      }

      acc0 = __builtin_amdgcn_wmma_f32_16x16x32_bf16(false, afrag, false, b0,
                                                     (short)0, acc0, false, false);
      acc1 = __builtin_amdgcn_wmma_f32_16x16x32_bf16(false, afrag, false, b1,
                                                     (short)0, acc1, false, false);
      __syncthreads();
    }
  }

  // epilogue: conv-bias + BN + ReLU, store bf16. D layout: M = v + 8*half, N = lane&15.
  const int pcol0 = pbase + Nbase + (lane & 15);
#pragma unroll
  for (int v = 0; v < 8; ++v) {
    int o = Mblock + Moff + v + 8 * hh;
    float ib = bias[o], ig = gam[o], ibe = bet[o], im = mu[o];
    float ir = rsqrtf(var[o] + 1e-5f);
    float s0 = fmaxf(ig * (acc0[v] + ib - im) * ir + ibe, 0.0f);
    float s1 = fmaxf(ig * (acc1[v] + ib - im) * ir + ibe, 0.0f);
    size_t base = ((size_t)img * CIN_ + o) * PIX_;
    Y[base + pcol0]      = f2bf(s0);
    Y[base + pcol0 + 16] = f2bf(s1);
  }
}

// ---------- 1x1 conv (256 -> 80) WMMA, wave-per-16x16 tile ----------
__global__ __launch_bounds__(256)
void conv1x1_wmma(const unsigned short* __restrict__ X,   // [NIMG][CIN][PIX] bf16
                  const unsigned short* __restrict__ Wa,  // [COUT][CIN] bf16
                  const float* __restrict__ bias,
                  float* __restrict__ feat) {             // [NIMG][COUT][PIX] f32
  const int lane = threadIdx.x & 31;
  const int wid  = threadIdx.x >> 5;
  const int id   = blockIdx.x * 8 + wid;       // NIMG * 5 * 176 tiles
  const int img  = id / (5 * 176);
  const int rem  = id % (5 * 176);
  const int mt   = rem / 176;
  const int pt   = rem % 176;
  const int m    = lane & 15;
  const int hh   = lane >> 4;
  const int pbase = pt * 16;

  v8f acc = {};
  const int o_row = mt * 16 + m;
#pragma unroll
  for (int kb = 0; kb < CIN_; kb += 32) {
    v16bf afrag;
    ((bf16x8*)&afrag)[0] = *(const bf16x8*)(Wa + (size_t)o_row * CIN_ + kb + 8 * hh);
    ((bf16x8*)&afrag)[1] = *(const bf16x8*)(Wa + (size_t)o_row * CIN_ + kb + 16 + 8 * hh);
    v16bf bfrag;
#pragma unroll
    for (int s = 0; s < 16; ++s) {
      int kr = (s < 8) ? (8 * hh + s) : (16 + 8 * hh + (s - 8));
      bfrag[s] = ((const __bf16*)X)[((size_t)img * CIN_ + kb + kr) * PIX_ + pbase + m];
    }
    acc = __builtin_amdgcn_wmma_f32_16x16x32_bf16(false, afrag, false, bfrag,
                                                  (short)0, acc, false, false);
  }
  const int pcol = pbase + (lane & 15);
#pragma unroll
  for (int v = 0; v < 8; ++v) {
    int o = mt * 16 + v + 8 * hh;
    feat[((size_t)img * COUT_ + o) * PIX_ + pcol] = acc[v] + bias[o];
  }
}

// ---------- projection precompute: per (cam, point) -> ix, iy, on_img ----------
__global__ void project_kernel(const float* __restrict__ l2i,
                               float* __restrict__ proj) {
  int i = blockIdx.x * blockDim.x + threadIdx.x;
  if (i >= NIMG_ * NPTS_) return;
  int cam = i / NPTS_;
  int pid = i % NPTS_;
  int xi = pid / (NY_ * NZ_);
  int r  = pid % (NY_ * NZ_);
  int yi = r / NZ_;
  int zi = r % NZ_;
  float x = 0.8f * xi - 50.8f;
  float y = 0.8f * yi - 50.8f;
  float z = 0.8f * zi - 4.6f;
  const float* M = l2i + cam * 16;
  float Xc = M[0] * x + M[1] * y + M[2]  * z + M[3];
  float Yc = M[4] * x + M[5] * y + M[6]  * z + M[7];
  float Zc = M[8] * x + M[9] * y + M[10] * z + M[11];
  float px = Xc / Zc, py = Yc / Zc;
  bool on = (px < IMGW_) && (px > 0.0f) && (py < IMGH_) && (py > 0.0f) && (Zc > 0.1f);
  float ix = px * ((float)(FW_ - 1) / IMGW_);  // (gx+1)*0.5*(W-1)
  float iy = py * ((float)(FH_ - 1) / IMGH_);
  float* P = proj + (size_t)i * 3;
  P[0] = ix; P[1] = iy; P[2] = on ? 1.0f : 0.0f;
}

// ---------- bilinear sample + max over cams + mean over z ----------
__global__ __launch_bounds__(256)
void bev_kernel(const float* __restrict__ feat,   // [NIMG][COUT][PIX]
                const float* __restrict__ proj,   // [NIMG][NPTS][3]
                float* __restrict__ out) {        // [COUT][NX][NY]
  int i = blockIdx.x * blockDim.x + threadIdx.x;
  if (i >= COUT_ * NX_ * NY_) return;
  int co = i / (NX_ * NY_);
  int xy = i % (NX_ * NY_);       // xy = xi*128 + yi;  pid = xy*10 + zi
  float accz = 0.0f;
  for (int zi = 0; zi < NZ_; ++zi) {
    int pid = xy * NZ_ + zi;
    float best = -3.4e38f;
    for (int cam = 0; cam < NIMG_; ++cam) {
      const float* P = proj + ((size_t)cam * NPTS_ + pid) * 3;
      float fl = P[2];
      float s = 0.0f;
      if (fl != 0.0f) {
        float ix = P[0], iy = P[1];
        float ix0 = floorf(ix), iy0 = floorf(iy);
        float wx1 = ix - ix0, wy1 = iy - iy0;
        int i0 = (int)ix0, j0 = (int)iy0;
        const float* F = feat + ((size_t)cam * COUT_ + co) * PIX_;
#pragma unroll
        for (int dj = 0; dj < 2; ++dj)
#pragma unroll
          for (int di = 0; di < 2; ++di) {
            int ii = i0 + di, jj = j0 + dj;
            float w = (di ? wx1 : 1.0f - wx1) * (dj ? wy1 : 1.0f - wy1);
            if ((unsigned)ii < (unsigned)FW_ && (unsigned)jj < (unsigned)FH_)
              s += F[jj * FW_ + ii] * w;
          }
      }
      best = fmaxf(best, s);   // masked cams contribute exactly 0, as in reference
    }
    accz += best;
  }
  out[i] = accz * 0.1f;   // mean over NZ
}

// ---------- workspace layout (bytes, 256-aligned) ----------
#define OFF_XBF   ((size_t)0)
#define OFF_Y1    ((size_t)8650752)
#define OFF_Y2    ((size_t)17301504)
#define OFF_W1A   ((size_t)25952256)
#define OFF_W2A   ((size_t)27131904)
#define OFF_W3A   ((size_t)28311552)
#define OFF_FEAT  ((size_t)28352512)
#define OFF_PROJ  ((size_t)33759232)

extern "C" void kernel_launch(void* const* d_in, const int* in_sizes, int n_in,
                              void* d_out, int out_size, void* d_ws, size_t ws_size,
                              hipStream_t stream) {
  (void)in_sizes; (void)n_in; (void)out_size; (void)ws_size;
  const float* img = (const float*)d_in[0];
  const float* l2i = (const float*)d_in[1];
  const float* w1  = (const float*)d_in[2];
  const float* b1  = (const float*)d_in[3];
  const float* g1  = (const float*)d_in[4];
  const float* be1 = (const float*)d_in[5];
  const float* m1  = (const float*)d_in[6];
  const float* v1  = (const float*)d_in[7];
  const float* w2  = (const float*)d_in[8];
  const float* b2  = (const float*)d_in[9];
  const float* g2  = (const float*)d_in[10];
  const float* be2 = (const float*)d_in[11];
  const float* m2  = (const float*)d_in[12];
  const float* v2  = (const float*)d_in[13];
  const float* w3  = (const float*)d_in[14];
  const float* b3  = (const float*)d_in[15];
  float* out = (float*)d_out;

  char* ws = (char*)d_ws;
  unsigned short* xbf  = (unsigned short*)(ws + OFF_XBF);
  unsigned short* y1bf = (unsigned short*)(ws + OFF_Y1);
  unsigned short* y2bf = (unsigned short*)(ws + OFF_Y2);
  unsigned short* w1a  = (unsigned short*)(ws + OFF_W1A);
  unsigned short* w2a  = (unsigned short*)(ws + OFF_W2A);
  unsigned short* w3a  = (unsigned short*)(ws + OFF_W3A);
  float* feat = (float*)(ws + OFF_FEAT);
  float* proj = (float*)(ws + OFF_PROJ);

  const int n_img = NIMG_ * CIN_ * PIX_;                  // 4,325,376
  cvt_bf16_kernel<<<(n_img + 255) / 256, 256, 0, stream>>>(img, xbf, n_img);
  pack_w3x3_kernel<<<(CIN_ * K3_ + 255) / 256, 256, 0, stream>>>(w1, w1a);
  pack_w3x3_kernel<<<(CIN_ * K3_ + 255) / 256, 256, 0, stream>>>(w2, w2a);
  cvt_bf16_kernel<<<(COUT_ * CIN_ + 255) / 256, 256, 0, stream>>>(w3, w3a, COUT_ * CIN_);

  dim3 cgrid(PIX_ / 64, CIN_ / 64, NIMG_);                // (44, 4, 6)
  conv3x3_bn_relu_wmma<<<cgrid, 256, 0, stream>>>(xbf,  w1a, b1, g1, be1, m1, v1, y1bf);
  conv3x3_bn_relu_wmma<<<cgrid, 256, 0, stream>>>(y1bf, w2a, b2, g2, be2, m2, v2, y2bf);

  conv1x1_wmma<<<(NIMG_ * 5 * 176) / 8, 256, 0, stream>>>(y2bf, w3a, b3, feat);

  project_kernel<<<(NIMG_ * NPTS_ + 255) / 256, 256, 0, stream>>>(l2i, proj);
  bev_kernel<<<(COUT_ * NX_ * NY_ + 255) / 256, 256, 0, stream>>>(feat, proj, out);
}
